// ChemResBlock_20521353740395
// MI455X (gfx1250) — matrix operand order repaired and checked by reading, not compiled, not run on hardware
//
#include <hip/hip_runtime.h>
#include <hip/hip_bf16.h>

typedef __bf16 bf16;
typedef __attribute__((ext_vector_type(16))) __bf16 v16bf;
typedef __attribute__((ext_vector_type(8)))  __bf16 v8bf;
typedef __attribute__((ext_vector_type(4)))  __bf16 v4bf;
typedef __attribute__((ext_vector_type(8)))  float  v8f;

#define NA 2048              // atoms
#define ND 64                // depth
#define NF 12                // filter length
#define NK (NA * NF)         // flattened K = 24576
#define NC (ND + 2)          // filter last dim = 66
#define KS 16                // split-K factor
#define KSLICE (NK / KS)     // 1536 K per slice
#define KB 64                // K chunk per LDS stage
#define MT 64                // M rows per workgroup (2 x 16 per wave)
#define LDA (KB + 8)         // padded LDS row (bf16) to spread banks

// ---- CDNA5 async global->LDS path (guarded; falls back to load+ds_store) ----
#if defined(__has_builtin)
#if __has_builtin(__builtin_amdgcn_global_load_async_to_lds_b128)
#define USE_ASYNC_LDS 1
#endif
#endif

#ifdef USE_ASYNC_LDS
typedef int v4i_vs __attribute__((vector_size(4 * sizeof(int))));
typedef __attribute__((address_space(1))) v4i_vs* gptr_v4i;
typedef __attribute__((address_space(3))) v4i_vs* lptr_v4i;

__device__ __forceinline__ void async_cp16(const bf16* g, bf16* l) {
    __builtin_amdgcn_global_load_async_to_lds_b128((gptr_v4i)g, (lptr_v4i)l, 0, 0);
}
__device__ __forceinline__ void async_wait_le4() {   // prior chunk (4 copies) done
#if __has_builtin(__builtin_amdgcn_s_wait_asynccnt)
    __builtin_amdgcn_s_wait_asynccnt(4);
#else
    asm volatile("s_wait_asynccnt 0x4" ::: "memory");
#endif
}
__device__ __forceinline__ void async_wait0() {
#if __has_builtin(__builtin_amdgcn_s_wait_asynccnt)
    __builtin_amdgcn_s_wait_asynccnt(0);
#else
    asm volatile("s_wait_asynccnt 0x0" ::: "memory");
#endif
}
#else
__device__ __forceinline__ void async_wait_le4() {}
__device__ __forceinline__ void async_wait0() {}
#endif

union Frag16 { v16bf v; v8bf h[2]; };

// ---- one-time f32 -> bf16 cast of conn (2048 x 24576) -----------------------
__global__ void k_cast_bf16(const float* __restrict__ src, bf16* __restrict__ dst, int n4) {
    int i = blockIdx.x * blockDim.x + threadIdx.x;
    if (i >= n4) return;
    float4 v = ((const float4*)src)[i];
    v4bf o = { (bf16)v.x, (bf16)v.y, (bf16)v.z, (bf16)v.w };
    *(v4bf*)(dst + 4 * (size_t)i) = o;
}

// ---- bias[a,o] = sum_f bond[a,f,0]*filt[o,f,64] + bond[a,f,1]*filt[o,f,65] --
__global__ void k_bias(const float* __restrict__ bond, const float* __restrict__ filt,
                       float* __restrict__ bias) {
    int a = blockIdx.x;
    int o = threadIdx.x;                 // 64 threads
    __shared__ float b[NF * 2];
    if (threadIdx.x < NF * 2) b[threadIdx.x] = bond[a * NF * 2 + threadIdx.x];
    __syncthreads();
    float s = 0.f;
#pragma unroll
    for (int f = 0; f < NF; ++f) {
        const float* w = filt + (o * NF + f) * NC + ND;
        s += b[f * 2 + 0] * w[0];
        s += b[f * 2 + 1] * w[1];
    }
    bias[a * ND + o] = s;
}

// ---- Zt[o, n*NF+f] = sum_d cur[n,d] * filt[o,f,d]   (stored o-major, bf16) --
__global__ void k_zt(const float* __restrict__ cur, const float* __restrict__ filt,
                     bf16* __restrict__ Zt) {
    int nf = blockIdx.x;                 // n*NF + f
    int n = nf / NF, f = nf % NF;
    int o = threadIdx.x;                 // 64 threads
    __shared__ float xr[ND];
    xr[o] = cur[n * ND + o];
    __syncthreads();
    const float* w = filt + (o * NF + f) * NC;
    float s = 0.f;
#pragma unroll
    for (int d = 0; d < ND; ++d) s += xr[d] * w[d];
    Zt[(size_t)o * NK + nf] = (bf16)s;
}

// ---- split-K bf16 GEMM: part[ks, a, o] = conn_bf16[a, kslice] @ Zt^T --------
__global__ __launch_bounds__(256)
void k_gemm(const bf16* __restrict__ Ab,   // (NA, NK) bf16
            const bf16* __restrict__ Bt,   // (ND, NK) bf16  (Z transposed)
            float* __restrict__ part) {    // (KS, NA, ND) f32
    __shared__ __attribute__((aligned(16))) bf16 As[2][MT][LDA];
    __shared__ __attribute__((aligned(16))) bf16 Bs[2][ND][LDA];

    const int tid   = threadIdx.x;
    const int wave  = tid >> 5;            // 8 waves (wave32)
    const int lane  = tid & 31;
    const int mhalf = (wave >> 2) * 32;    // 0 or 32: this wave's 32-row band
    const int ntile = (wave & 3) * 16;     // 0,16,32,48
    const int a0     = blockIdx.x * MT;
    const int k0base = blockIdx.y * KSLICE;

    const int lm    = lane & 15;
    const int ahalf = (lane < 16) ? 0 : 8;   // ISA 16-bit A 16x32 lane layout
    const int bhalf = (lane < 16) ? 0 : 8;   // ISA 16-bit B 32x16 lane layout
    const int mofs  = (lane < 16) ? 0 : 8;   // ISA f32 C/D 16x16 lane layout

    const int arow = tid >> 2;             // 0..63 (one row per 4 threads)
    const int acol = (tid & 3) * 8;        // bf16 offset of 16B granule

    v8f acc0 = {};
    v8f acc1 = {};

    // stage one KB-chunk of A (64x64) and B (64x64) into LDS buffer `buf`
    auto issue = [&](int kt, int buf) {
        const bf16* ga = &Ab[(size_t)(a0 + arow) * NK + k0base + kt + acol];
        const bf16* gb = &Bt[(size_t)arow        * NK + k0base + kt + acol];
#ifdef USE_ASYNC_LDS
        async_cp16(ga,      &As[buf][arow][acol]);
        async_cp16(ga + 32, &As[buf][arow][acol + 32]);
        async_cp16(gb,      &Bs[buf][arow][acol]);
        async_cp16(gb + 32, &Bs[buf][arow][acol + 32]);
#else
        *(uint4*)&As[buf][arow][acol]      = *(const uint4*)ga;
        *(uint4*)&As[buf][arow][acol + 32] = *(const uint4*)(ga + 32);
        *(uint4*)&Bs[buf][arow][acol]      = *(const uint4*)gb;
        *(uint4*)&Bs[buf][arow][acol + 32] = *(const uint4*)(gb + 32);
#endif
    };

    issue(0, 0);                           // prologue: chunk 0 -> buffer 0
    int buf = 0;
    for (int kt = 0; kt < KSLICE; kt += KB, buf ^= 1) {
        if (kt + KB < KSLICE) {
            issue(kt + KB, buf ^ 1);       // stream next chunk behind compute
            async_wait_le4();              // current chunk's 4 copies complete
        } else {
            async_wait0();
        }
        __syncthreads();                   // chunk visible to all waves

        if (kt + 2 * KB < KSLICE)
            __builtin_prefetch(&Ab[(size_t)(a0 + arow) * NK + k0base + kt + 2 * KB + acol], 0, 1);

#pragma unroll
        for (int kk = 0; kk < KB; kk += 32) {
            Frag16 af0, af1, bfr;
            // B frag: lane holds column N=lm, K kk+bhalf..+7 and kk+bhalf+16..+23
            bfr.h[0] = *(const v8bf*)&Bs[buf][ntile + lm][kk + bhalf];
            bfr.h[1] = *(const v8bf*)&Bs[buf][ntile + lm][kk + bhalf + 16];
            // A frags: lane holds M=lm of each 16-row subtile
            af0.h[0] = *(const v8bf*)&As[buf][mhalf + lm][kk + ahalf];
            af0.h[1] = *(const v8bf*)&As[buf][mhalf + lm][kk + ahalf + 16];
            af1.h[0] = *(const v8bf*)&As[buf][mhalf + 16 + lm][kk + ahalf];
            af1.h[1] = *(const v8bf*)&As[buf][mhalf + 16 + lm][kk + ahalf + 16];
            acc0 = __builtin_amdgcn_wmma_f32_16x16x32_bf16(
                false, af0.v, false, bfr.v, (short)0, acc0, false, false);
            acc1 = __builtin_amdgcn_wmma_f32_16x16x32_bf16(
                false, af1.v, false, bfr.v, (short)0, acc1, false, false);
        }
        __syncthreads();                   // done reading `buf` before overwrite
    }

    float* p0 = part + ((size_t)blockIdx.y * NA + a0 + mhalf + mofs) * ND + ntile + lm;
    float* p1 = p0 + (size_t)16 * ND;
#pragma unroll
    for (int r = 0; r < 8; ++r) {
        p0[(size_t)r * ND] = acc0[r];
        p1[(size_t)r * ND] = acc1[r];
    }
}

// ---- epilogue: out = relu( sum_ks part + bias [+ resid] ) -------------------
__global__ void k_epi(const float* __restrict__ part, const float* __restrict__ bias,
                      const float* __restrict__ resid, float* __restrict__ out) {
    int i = blockIdx.x * blockDim.x + threadIdx.x;   // over NA*ND
    float s = bias[i];
#pragma unroll
    for (int ks = 0; ks < KS; ++ks) s += part[(size_t)ks * NA * ND + i];
    if (resid) s += resid[i];
    out[i] = s > 0.f ? s : 0.f;
}

extern "C" void kernel_launch(void* const* d_in, const int* in_sizes, int n_in,
                              void* d_out, int out_size, void* d_ws, size_t ws_size,
                              hipStream_t stream) {
    (void)in_sizes; (void)n_in; (void)out_size; (void)ws_size;
    const float* x    = (const float*)d_in[0];   // (2048, 64)
    const float* conn = (const float*)d_in[1];   // (2048, 2048, 12)
    const float* bond = (const float*)d_in[2];   // (2048, 12, 2)
    const float* f0   = (const float*)d_in[3];   // (64, 12, 66)
    const float* f1   = (const float*)d_in[4];
    float* out = (float*)d_out;

    char* ws = (char*)d_ws;
    auto alloc = [&](size_t bytes) {
        char* p = ws;
        ws += (bytes + 255) & ~(size_t)255;
        return p;
    };
    bf16*  connb = (bf16*) alloc((size_t)NA * NK * sizeof(bf16));   // 96 MiB
    bf16*  Zt    = (bf16*) alloc((size_t)ND * NK * sizeof(bf16));   // 3 MiB
    float* part  = (float*)alloc((size_t)KS * NA * ND * sizeof(float)); // 8 MiB
    float* bias0 = (float*)alloc((size_t)NA * ND * sizeof(float));
    float* bias1 = (float*)alloc((size_t)NA * ND * sizeof(float));
    float* tA    = (float*)alloc((size_t)NA * ND * sizeof(float));
    float* tB    = (float*)alloc((size_t)NA * ND * sizeof(float));

    // one-time conn -> bf16
    {
        int n4 = NA * NK / 4;
        k_cast_bf16<<<(n4 + 255) / 256, 256, 0, stream>>>(conn, connb, n4);
    }
    // per-filter-set constant bias terms
    k_bias<<<NA, ND, 0, stream>>>(bond, f0, bias0);
    k_bias<<<NA, ND, 0, stream>>>(bond, f1, bias1);

    const float* filts[2]  = { f0, f1 };
    float*       biases[2] = { bias0, bias1 };
    const dim3 gemm_grid(NA / MT, KS);

    const float* cur = x;
    for (int l = 0; l < 2; ++l) {
        // application 1: out = relu(chemconv(cur))
        k_zt<<<NK, ND, 0, stream>>>(cur, filts[l], Zt);
        k_gemm<<<gemm_grid, 256, 0, stream>>>(connb, Zt, part);
        k_epi<<<NA * ND / 256, 256, 0, stream>>>(part, biases[l], nullptr, tA);
        // application 2: out = relu(chemconv(out) + x)
        k_zt<<<NK, ND, 0, stream>>>(tA, filts[l], Zt);
        k_gemm<<<gemm_grid, 256, 0, stream>>>(connb, Zt, part);
        float* dst = (l == 1) ? out : tB;
        k_epi<<<NA * ND / 256, 256, 0, stream>>>(part, biases[l], x, dst);
        cur = dst;
    }
}
